// ViTMultiHeadAttention_85504208928924
// MI455X (gfx1250) — matrix operand order repaired and checked by reading, not compiled, not run on hardware
//
#include <hip/hip_runtime.h>

// ---------------------------------------------------------------------------
// ViT multi-head attention forward for MI455X (gfx1250, wave32, WMMA bf16)
//   B=16, T=1024, C=1024, H=16, Dh=64
// Pipeline: [1] qkv = x@w_qkv + b  (bf16 WMMA, scatter Q*scale,K,V^T to ws)
//           [2] flash attention per (b,h,qtile)  (bf16 WMMA, online softmax,
//               async global->LDS staging via GLOBAL_LOAD_ASYNC_TO_LDS_B128)
//           [3] out = y@w_out + b_out  (bf16 WMMA, fp32 store)
// ---------------------------------------------------------------------------

#define BATCH 16
#define TOK   1024
#define CH    1024
#define NHEAD 16
#define DHEAD 64
#define MTOT  (BATCH * TOK)     // 16384

typedef __attribute__((ext_vector_type(16))) __bf16 v16bf;
typedef __attribute__((ext_vector_type(8)))  __bf16 v8bf;
typedef __attribute__((ext_vector_type(8)))  float  v8f;
typedef int v4i __attribute__((vector_size(16)));

#define AS1 __attribute__((address_space(1)))
#define AS3 __attribute__((address_space(3)))

#if defined(__has_builtin)
#if __has_builtin(__builtin_amdgcn_global_load_async_to_lds_b128) && \
    __has_builtin(__builtin_amdgcn_s_wait_asynccnt)
#define HAVE_ASYNC_LDS 1
#endif
#endif

#ifdef HAVE_ASYNC_LDS
// Builtin signature (from clang diagnostic): (v4i AS1* src, v4i AS3* dst,
// imm offset, imm cpol). flat->LDS address is a low-32-bit truncate;
// flat->global is bit-identical, so integer casts construct the AS pointers.
#define ASYNC_CP16(lp, gp)                                          \
  __builtin_amdgcn_global_load_async_to_lds_b128(                   \
      (AS1 v4i*)(size_t)(gp), (AS3 v4i*)(unsigned)(size_t)(lp), 0, 0)
#endif

// ---- WMMA fragment loaders (CDNA5 ISA 7.12.2 16-bit layouts) ---------------
// A 16x32 bf16: lane L(0-15): M=L, K in {0..7,16..23}; lane L+16: K in {8..15,24..31}
__device__ __forceinline__ v16bf load_a_frag(const __bf16* base, int ldk) {
  const int lane = threadIdx.x & 31;
  const int row  = lane & 15;
  const int kb   = (lane >> 4) << 3;               // 0 or 8
  const __bf16* p = base + row * ldk + kb;
  v8bf lo = *(const v8bf*)p;                       // K = kb .. kb+7
  v8bf hi = *(const v8bf*)(p + 16);                // K = kb+16 .. kb+23
  v16bf r;
#pragma unroll
  for (int j = 0; j < 8; ++j) { r[j] = lo[j]; r[j + 8] = hi[j]; }
  return r;
}

// B 32x16 bf16 given B^T tile (n-major, K contiguous): lane L(0-15): N=L, K in {0..7,16..23};
// lane L+16: N=L, K in {8..15,24..31}
__device__ __forceinline__ v16bf load_b_frag(const __bf16* baseT, int ldk) {
  const int lane = threadIdx.x & 31;
  const int n    = lane & 15;
  const int kb   = (lane >> 4) << 3;               // 0 or 8
  const __bf16* p = baseT + n * ldk + kb;
  v8bf lo = *(const v8bf*)p;
  v8bf hi = *(const v8bf*)(p + 16);
  v16bf r;
#pragma unroll
  for (int j = 0; j < 8; ++j) { r[j] = lo[j]; r[j + 8] = hi[j]; }
  return r;
}

__device__ __forceinline__ v8f wmma_bf16(v16bf a, v16bf b, v8f c) {
  return __builtin_amdgcn_wmma_f32_16x16x32_bf16(false, a, false, b, (short)0, c,
                                                 false, false);
}

// ---------------------------------------------------------------------------
// Kernel 1: qkv = x @ w_qkv + b_qkv ; scatter to Q(scaled), K, V^T bf16 ws
// Grid: (3072/128, 16384/128), block 256 (8 waves), tile 128x128x32
// ---------------------------------------------------------------------------
__global__ __launch_bounds__(256) void k_qkv(
    const float* __restrict__ X, const float* __restrict__ W,
    const float* __restrict__ bias, __bf16* __restrict__ Qb,
    __bf16* __restrict__ Kb, __bf16* __restrict__ Vt) {
  __shared__ __bf16 As[128][40];   // [m][k], padded
  __shared__ __bf16 Bs[128][40];   // [n][k] (W transposed), padded

  const int tid  = threadIdx.x;
  const int lane = tid & 31;
  const int wv   = tid >> 5;
  const int wm   = (wv >> 2) * 64;     // wave m-offset in tile: 0/64
  const int wn   = (wv & 3) * 32;      // wave n-offset in tile: 0/32/64/96
  const int m0   = blockIdx.y * 128;
  const int n0   = blockIdx.x * 128;
  const int ln = lane & 15, hi = lane >> 4;

  v8f acc[4][2];
#pragma unroll
  for (int i = 0; i < 4; ++i)
#pragma unroll
    for (int j = 0; j < 2; ++j) acc[i][j] = (v8f){};

  for (int kt = 0; kt < CH; kt += 32) {
    const float* pA = X + (size_t)m0 * CH + kt;
    const float* pB = W + (size_t)kt * 3072 + n0;
    // Batch all global loads into registers first (one wait, 8 loads in
    // flight), then convert to bf16 and stage into LDS.
    float4 va[4], vb[4];
#pragma unroll
    for (int it = 0; it < 4; ++it) {
      int idx = tid + it * 256;
      va[it] = *(const float4*)(pA + (size_t)(idx >> 3) * CH + (idx & 7) * 4);
      vb[it] = *(const float4*)(pB + (size_t)(idx >> 5) * 3072 + (idx & 31) * 4);
    }
    if (kt + 32 < CH)  // global_prefetch_b8 of next A tile
      __builtin_prefetch(pA + (size_t)(tid >> 1) * CH + 32 + (tid & 1) * 16, 0, 1);
#pragma unroll
    for (int it = 0; it < 4; ++it) {
      int idx = tid + it * 256;
      int r = idx >> 3, c = (idx & 7) * 4;
      As[r][c + 0] = (__bf16)va[it].x; As[r][c + 1] = (__bf16)va[it].y;
      As[r][c + 2] = (__bf16)va[it].z; As[r][c + 3] = (__bf16)va[it].w;
      int kk = idx >> 5, nn = (idx & 31) * 4;
      Bs[nn + 0][kk] = (__bf16)vb[it].x; Bs[nn + 1][kk] = (__bf16)vb[it].y;
      Bs[nn + 2][kk] = (__bf16)vb[it].z; Bs[nn + 3][kk] = (__bf16)vb[it].w;
    }
    __syncthreads();

    v16bf a[4], b[2];
#pragma unroll
    for (int i = 0; i < 4; ++i) a[i] = load_a_frag(&As[wm + i * 16][0], 40);
#pragma unroll
    for (int j = 0; j < 2; ++j) b[j] = load_b_frag(&Bs[wn + j * 16][0], 40);
#pragma unroll
    for (int i = 0; i < 4; ++i)
#pragma unroll
      for (int j = 0; j < 2; ++j) acc[i][j] = wmma_bf16(a[i], b[j], acc[i][j]);
    __syncthreads();
  }

  // Epilogue: bias add + scatter into attention-friendly bf16 layouts
#pragma unroll
  for (int i = 0; i < 4; ++i)
#pragma unroll
    for (int j = 0; j < 2; ++j) {
      const int n = n0 + wn + j * 16 + ln;
      const float bval = bias[n];
#pragma unroll
      for (int r = 0; r < 8; ++r) {
        const int m = m0 + wm + i * 16 + r + 8 * hi;
        const float v = acc[i][j][r] + bval;
        const int bb = m >> 10, t = m & 1023;
        if (n < CH) {                               // Q (fold softmax scale)
          const int h = n >> 6, d = n & 63;
          Qb[(((size_t)bb * NHEAD + h) * TOK + t) * DHEAD + d] =
              (__bf16)(v * 0.125f);
        } else if (n < 2 * CH) {                    // K
          const int h = (n - CH) >> 6, d = (n - CH) & 63;
          Kb[(((size_t)bb * NHEAD + h) * TOK + t) * DHEAD + d] = (__bf16)v;
        } else {                                    // V stored transposed
          const int h = (n - 2 * CH) >> 6, d = (n - 2 * CH) & 63;
          Vt[(((size_t)bb * NHEAD + h) * DHEAD + d) * TOK + t] = (__bf16)v;
        }
      }
    }
}

// ---------------------------------------------------------------------------
// Kernel 2: flash attention. Grid (T/64, B*H), block 128 (4 waves x 16 rows)
// ---------------------------------------------------------------------------
__global__ __launch_bounds__(128) void k_attn(
    const __bf16* __restrict__ Qg, const __bf16* __restrict__ Kg,
    const __bf16* __restrict__ Vg, __bf16* __restrict__ Yg) {
  __shared__ __bf16 Kt[64][64];        // Q staging, then key block [key][d]
  __shared__ __bf16 Vtt[64][64];       // V^T block [d][key]
  __shared__ float  Sf[4][16][64];     // per-wave raw scores
  __shared__ __bf16 Pb[4][16][64];     // per-wave P (bf16, A-operand layout)
  __shared__ float  rowM[64], rowL[64], rowA[64];

  const int tid = threadIdx.x, lane = tid & 31, wv = tid >> 5;
  const int ln = lane & 15, hi = lane >> 4;
  const int bh = blockIdx.y;           // b*NHEAD + h
  const int q0 = blockIdx.x * 64;
  const __bf16* Q = Qg + (size_t)bh * TOK * DHEAD;
  const __bf16* K = Kg + (size_t)bh * TOK * DHEAD;
  const __bf16* V = Vg + (size_t)bh * DHEAD * TOK;

  // Stage Q tile (async LDS copy when available), hoist Q frags to registers
#pragma unroll
  for (int it = 0; it < 4; ++it) {
    int idx = tid + it * 128;
    int r = idx >> 3, s = (idx & 7) * 8;
#ifdef HAVE_ASYNC_LDS
    ASYNC_CP16(&Kt[r][s], Q + (size_t)(q0 + r) * DHEAD + s);
#else
    *(v8bf*)&Kt[r][s] = *(const v8bf*)(Q + (size_t)(q0 + r) * DHEAD + s);
#endif
  }
  if (tid < 64) { rowM[tid] = -1e30f; rowL[tid] = 0.0f; }
#ifdef HAVE_ASYNC_LDS
  __builtin_amdgcn_s_wait_asynccnt(0);
#endif
  __syncthreads();
  v16bf qa0 = load_a_frag(&Kt[wv * 16][0], 64);
  v16bf qa1 = load_a_frag(&Kt[wv * 16][32], 64);
  __syncthreads();

  v8f o[4];
#pragma unroll
  for (int d = 0; d < 4; ++d) o[d] = (v8f){};

  for (int k0 = 0; k0 < TOK; k0 += 64) {
    // Stage 64-key K block and V^T block
#pragma unroll
    for (int it = 0; it < 4; ++it) {
      int idx = tid + it * 128;
      int r = idx >> 3, s = (idx & 7) * 8;
#ifdef HAVE_ASYNC_LDS
      ASYNC_CP16(&Kt[r][s], K + (size_t)(k0 + r) * DHEAD + s);
      ASYNC_CP16(&Vtt[r][s], V + (size_t)r * TOK + k0 + s);
#else
      *(v8bf*)&Kt[r][s]  = *(const v8bf*)(K + (size_t)(k0 + r) * DHEAD + s);
      *(v8bf*)&Vtt[r][s] = *(const v8bf*)(V + (size_t)r * TOK + k0 + s);
#endif
    }
#ifdef HAVE_ASYNC_LDS
    __builtin_amdgcn_s_wait_asynccnt(0);
#endif
    __syncthreads();

    // S = Q K^T  (16x64 per wave, f32)
    v8f s4[4];
#pragma unroll
    for (int ns = 0; ns < 4; ++ns) {
      v8f sa = (v8f){};
      sa = wmma_bf16(qa0, load_b_frag(&Kt[ns * 16][0], 64), sa);
      sa = wmma_bf16(qa1, load_b_frag(&Kt[ns * 16][32], 64), sa);
      s4[ns] = sa;
    }
#pragma unroll
    for (int ns = 0; ns < 4; ++ns)
#pragma unroll
      for (int r = 0; r < 8; ++r)
        Sf[wv][r + 8 * hi][ns * 16 + ln] = s4[ns][r];
    asm volatile("s_wait_dscnt 0" ::: "memory");

    // Online softmax: each half-wave lane owns 32 columns of one row
    {
      const int tr = wv * 16 + ln;
      float mloc = -1e30f;
#pragma unroll
      for (int c = 0; c < 32; ++c) mloc = fmaxf(mloc, Sf[wv][ln][hi * 32 + c]);
      mloc = fmaxf(mloc, __shfl_xor(mloc, 16, 32));
      const float mold = rowM[tr];
      const float mnew = fmaxf(mold, mloc);
      const float alpha = __expf(mold - mnew);
      float ssum = 0.0f;
#pragma unroll
      for (int c = 0; c < 32; ++c) {
        float p = __expf(Sf[wv][ln][hi * 32 + c] - mnew);
        Pb[wv][ln][hi * 32 + c] = (__bf16)p;
        ssum += p;
      }
      ssum += __shfl_xor(ssum, 16, 32);
      if (hi == 0) {
        rowL[tr] = alpha * rowL[tr] + ssum;
        rowM[tr] = mnew;
        rowA[tr] = alpha;
      }
    }
    asm volatile("s_wait_dscnt 0" ::: "memory");

    // Rescale running O by per-row alpha
    {
      const float4 a0 = *(const float4*)&rowA[wv * 16 + 8 * hi];
      const float4 a1 = *(const float4*)&rowA[wv * 16 + 8 * hi + 4];
      const float av[8] = {a0.x, a0.y, a0.z, a0.w, a1.x, a1.y, a1.z, a1.w};
#pragma unroll
      for (int d = 0; d < 4; ++d)
#pragma unroll
        for (int r = 0; r < 8; ++r) o[d][r] *= av[r];
    }

    // O += P V  (A = P bf16, B^T = Vtt)
#pragma unroll
    for (int d = 0; d < 4; ++d) {
      o[d] = wmma_bf16(load_a_frag(&Pb[wv][0][0], 64),
                       load_b_frag(&Vtt[d * 16][0], 64), o[d]);
      o[d] = wmma_bf16(load_a_frag(&Pb[wv][0][32], 64),
                       load_b_frag(&Vtt[d * 16][32], 64), o[d]);
    }
    __syncthreads();
  }

  // Normalize by row sums, store Y in [B][T][H*Dh] bf16
  const float4 l0 = *(const float4*)&rowL[wv * 16 + 8 * hi];
  const float4 l1 = *(const float4*)&rowL[wv * 16 + 8 * hi + 4];
  const float lv[8] = {l0.x, l0.y, l0.z, l0.w, l1.x, l1.y, l1.z, l1.w};
  const int b = bh >> 4, h = bh & 15;
#pragma unroll
  for (int d = 0; d < 4; ++d)
#pragma unroll
    for (int r = 0; r < 8; ++r) {
      const int t = q0 + wv * 16 + r + 8 * hi;
      const int col = h * DHEAD + d * 16 + ln;
      Yg[((size_t)b * TOK + t) * CH + col] = (__bf16)(o[d][r] / lv[r]);
    }
}

// ---------------------------------------------------------------------------
// Kernel 3: out = Y @ w_out + b_out (fp32 store). Grid (1024/128, 16384/128)
// ---------------------------------------------------------------------------
__global__ __launch_bounds__(256) void k_out(
    const __bf16* __restrict__ Yg, const float* __restrict__ W,
    const float* __restrict__ bias, float* __restrict__ Out) {
  __shared__ __bf16 As[128][40];
  __shared__ __bf16 Bs[128][40];

  const int tid  = threadIdx.x;
  const int lane = tid & 31;
  const int wv   = tid >> 5;
  const int wm   = (wv >> 2) * 64;
  const int wn   = (wv & 3) * 32;
  const int m0   = blockIdx.y * 128;
  const int n0   = blockIdx.x * 128;
  const int ln = lane & 15, hi = lane >> 4;

  v8f acc[4][2];
#pragma unroll
  for (int i = 0; i < 4; ++i)
#pragma unroll
    for (int j = 0; j < 2; ++j) acc[i][j] = (v8f){};

  for (int kt = 0; kt < CH; kt += 32) {
    const __bf16* pA = Yg + (size_t)m0 * CH + kt;
    const float*  pB = W + (size_t)kt * CH + n0;
    v8bf ya[2];
    float4 vb[4];
#pragma unroll
    for (int it = 0; it < 2; ++it) {
      int idx = tid + it * 256;
      ya[it] = *(const v8bf*)(pA + (size_t)(idx >> 2) * CH + (idx & 3) * 8);
    }
#pragma unroll
    for (int it = 0; it < 4; ++it) {
      int idx = tid + it * 256;
      vb[it] = *(const float4*)(pB + (size_t)(idx >> 5) * CH + (idx & 31) * 4);
    }
    if (kt + 32 < CH)
      __builtin_prefetch(pB + (size_t)(tid >> 3) * CH + 32 * CH + (tid & 7) * 16,
                         0, 1);
#pragma unroll
    for (int it = 0; it < 2; ++it) {
      int idx = tid + it * 256;
      *(v8bf*)&As[idx >> 2][(idx & 3) * 8] = ya[it];
    }
#pragma unroll
    for (int it = 0; it < 4; ++it) {
      int idx = tid + it * 256;
      int kk = idx >> 5, nn = (idx & 31) * 4;
      Bs[nn + 0][kk] = (__bf16)vb[it].x; Bs[nn + 1][kk] = (__bf16)vb[it].y;
      Bs[nn + 2][kk] = (__bf16)vb[it].z; Bs[nn + 3][kk] = (__bf16)vb[it].w;
    }
    __syncthreads();

    v16bf a[4], b[2];
#pragma unroll
    for (int i = 0; i < 4; ++i) a[i] = load_a_frag(&As[wm + i * 16][0], 40);
#pragma unroll
    for (int j = 0; j < 2; ++j) b[j] = load_b_frag(&Bs[wn + j * 16][0], 40);
#pragma unroll
    for (int i = 0; i < 4; ++i)
#pragma unroll
      for (int j = 0; j < 2; ++j) acc[i][j] = wmma_bf16(a[i], b[j], acc[i][j]);
    __syncthreads();
  }

#pragma unroll
  for (int i = 0; i < 4; ++i)
#pragma unroll
    for (int j = 0; j < 2; ++j) {
      const int n = n0 + wn + j * 16 + ln;
      const float bval = bias[n];
#pragma unroll
      for (int r = 0; r < 8; ++r) {
        const int m = m0 + wm + i * 16 + r + 8 * hi;
        Out[(size_t)m * CH + n] = acc[i][j][r] + bval;
      }
    }
}

// ---------------------------------------------------------------------------
extern "C" void kernel_launch(void* const* d_in, const int* in_sizes, int n_in,
                              void* d_out, int out_size, void* d_ws,
                              size_t ws_size, hipStream_t stream) {
  (void)in_sizes; (void)n_in; (void)out_size; (void)ws_size;
  const float* x     = (const float*)d_in[0];
  const float* w_qkv = (const float*)d_in[1];
  const float* b_qkv = (const float*)d_in[2];
  const float* w_out = (const float*)d_in[3];
  const float* b_out = (const float*)d_in[4];

  // bf16 workspace: Q | K | V^T | Y  (4 x 32 MB = 128 MB)
  __bf16* Qb = (__bf16*)d_ws;
  __bf16* Kb = Qb + (size_t)MTOT * CH;       // 16,777,216 elems each
  __bf16* Vt = Kb + (size_t)MTOT * CH;
  __bf16* Yb = Vt + (size_t)MTOT * CH;

  k_qkv<<<dim3(3 * CH / 128, MTOT / 128), 256, 0, stream>>>(x, w_qkv, b_qkv,
                                                            Qb, Kb, Vt);
  k_attn<<<dim3(TOK / 64, BATCH * NHEAD), 128, 0, stream>>>(Qb, Kb, Vt, Yb);
  k_out<<<dim3(CH / 128, MTOT / 128), 256, 0, stream>>>(Yb, w_out, b_out,
                                                        (float*)d_out);
}
